// EditDecoderRNN_18296560681100
// MI455X (gfx1250) — compile-verified, matrix operand order checked by reading, not run on hardware
//
#include <hip/hip_runtime.h>

#define B_  32
#define T_  100
#define S_  100
#define H_  512
#define E_  256
#define V_  30000
#define VP_ 30080
#define G4_ 2048        // 4*H
#define NBLK_CHAIN 32u  // blocks per chain in the persistent recurrence
#define LDH 520         // padded LDS row stride in bf16 (260 dwords = 4 mod 64 banks)

typedef __bf16 bf16_t;
typedef __attribute__((ext_vector_type(8)))  __bf16 v8bf;
typedef __attribute__((ext_vector_type(16))) __bf16 v16bf;
typedef __attribute__((ext_vector_type(8)))  float  v8f;

// ---------------------------------------------------------------------------
// Fragment loader: 16x32 bf16 tile operand, K-contiguous rows in memory.
// Lane L<16 holds row (row0+L), K = {k0..k0+7} U {k0+16..k0+23};
// lane L+16 holds row (row0+L), K = {k0+8..k0+15} U {k0+24..k0+31}.
// Matches the CDNA5 ISA 16-bit A-matrix layout; B (stored [N,K]) uses the
// mirrored layout so the same loader serves both operands.
// ---------------------------------------------------------------------------
__device__ __forceinline__ v16bf load_frag(const bf16_t* __restrict__ base,
                                           int ld, int row0, int maxrow,
                                           int k0, int lane) {
  int r = row0 + (lane & 15);
  r = (r < maxrow) ? r : maxrow;
  int hi = (lane >> 4) & 1;
  const bf16_t* p = base + (size_t)r * ld + k0 + hi * 8;
  v8bf a = *(const v8bf*)p;
  v8bf b = *(const v8bf*)(p + 16);
  return __builtin_shufflevector(a, b, 0,1,2,3,4,5,6,7,8,9,10,11,12,13,14,15);
}

__device__ __forceinline__ v8f wmma_bf16(v16bf a, v16bf b, v8f c) {
  return __builtin_amdgcn_wmma_f32_16x16x32_bf16(false, a, false, b,
                                                 (short)0, c, false, false);
}

// ---------------------------------------------------------------------------
// Generic bf16 WMMA GEMM:  C[M,N] = act(A[M,K] * B[N,K]^T + bias)
// Block = 256 threads = 8 waves, block tile 128(M) x 128(N).
// Wave tile 32(M) x 64(N) = 2x4 WMMA tiles: 6 fragment loads -> 8 WMMAs.
// ---------------------------------------------------------------------------
template<bool HAS_BIAS, bool TANH_ACT, bool OUT_F32, bool OUT_BF16>
__global__ __launch_bounds__(256) void gemm_bf16_kernel(
    const bf16_t* __restrict__ A, const bf16_t* __restrict__ Bm,
    float* __restrict__ C, bf16_t* __restrict__ Cbf,
    const float* __restrict__ bias,
    int M, int K, int ldc, int Nvalid) {
  int wave = threadIdx.x >> 5;
  int lane = threadIdx.x & 31;
  int wm = wave & 3, wn = wave >> 2;
  int m0 = blockIdx.y * 128 + wm * 32;
  int n0 = blockIdx.x * 128 + wn * 64;

  v8f acc[2][4] = {};
  for (int k = 0; k < K; k += 32) {
    if (k + 32 < K) {  // wave-uniform; keeps EXEC all-ones through WMMA
      __builtin_prefetch(A + (size_t)m0 * K + k + 32, 0, 0);
      __builtin_prefetch(Bm + (size_t)n0 * K + k + 32, 0, 0);
    }
    v16bf a0 = load_frag(A,  K, m0,      M - 1,   k, lane);
    v16bf a1 = load_frag(A,  K, m0 + 16, M - 1,   k, lane);
    v16bf b0 = load_frag(Bm, K, n0,      1 << 28, k, lane);
    v16bf b1 = load_frag(Bm, K, n0 + 16, 1 << 28, k, lane);
    v16bf b2 = load_frag(Bm, K, n0 + 32, 1 << 28, k, lane);
    v16bf b3 = load_frag(Bm, K, n0 + 48, 1 << 28, k, lane);
    acc[0][0] = wmma_bf16(a0, b0, acc[0][0]);
    acc[0][1] = wmma_bf16(a0, b1, acc[0][1]);
    acc[0][2] = wmma_bf16(a0, b2, acc[0][2]);
    acc[0][3] = wmma_bf16(a0, b3, acc[0][3]);
    acc[1][0] = wmma_bf16(a1, b0, acc[1][0]);
    acc[1][1] = wmma_bf16(a1, b1, acc[1][1]);
    acc[1][2] = wmma_bf16(a1, b2, acc[1][2]);
    acc[1][3] = wmma_bf16(a1, b3, acc[1][3]);
  }

  int col  = lane & 15;
  int radd = (lane < 16) ? 0 : 8;
  for (int mt = 0; mt < 2; ++mt)
    for (int nt = 0; nt < 4; ++nt) {
      v8f a = acc[mt][nt];
      int gn = n0 + nt * 16 + col;
      for (int r = 0; r < 8; ++r) {
        int gm = m0 + mt * 16 + r + radd;
        if (gm < M && gn < Nvalid) {
          float v = a[r];
          if (HAS_BIAS) v += bias[gn];
          if (TANH_ACT) v = tanhf(v);
          if (OUT_F32)  C[(size_t)gm * ldc + gn] = v;
          if (OUT_BF16) Cbf[(size_t)gm * ldc + gn] = (bf16_t)v;
        }
      }
    }
}

// ---------------------------------------------------------------------------
// Persistent fused LSTM (both chains), all T_ steps in ONE launch.
// 64 blocks: blockIdx.x = 16-wide hidden strip, blockIdx.y = chain.
// Each wave preloads its 16x512 Wh strip into registers ONCE (16 fragments,
// held across all T steps). Per step, the whole h_prev (32KB) is staged into
// LDS with a batched cooperative copy (8 simultaneously-live uint4 per
// thread -> one L2 round trip), then the 16 A-fragments come from LDS via
// ds_load_b128 pairs (row stride padded to 520 bf16 to avoid bank conflicts).
// Inter-step sync: per-chain device-scope arrival counter (32 blocks).
// h is ping-pong double-buffered across steps.
// ---------------------------------------------------------------------------
__global__ __launch_bounds__(256) void lstm_persistent_kernel(
    const bf16_t* __restrict__ Wh_e, const bf16_t* __restrict__ Wh_w,
    const float* __restrict__ Gx_e,  const float* __restrict__ Gx_w,
    bf16_t* __restrict__ hbuf,       // [2 bufs][2 chains][B_*H_]
    float*  __restrict__ cstate,     // [2 chains][B_*H_]
    float*  __restrict__ out_e, float* __restrict__ out_w,
    bf16_t* __restrict__ out_e_bf,
    unsigned* __restrict__ bar) {    // [2 chains][T_] zeroed arrival counters
  __shared__ float  lds[8 * 256];
  __shared__ bf16_t sh_h[B_ * LDH];  // staged h_prev, padded rows
  int chain = blockIdx.y;
  const bf16_t* Wh = chain ? Wh_w : Wh_e;
  const float*  Gx = chain ? Gx_w : Gx_e;
  float* cs   = cstate + (size_t)chain * (B_ * H_);
  float* outp = chain ? out_w : out_e;
  unsigned* cbar = bar + (size_t)chain * T_;

  int wave = threadIdx.x >> 5, lane = threadIdx.x & 31;
  int mt = wave >> 2, gate = wave & 3;
  int m0 = mt * 16;
  int n0 = gate * H_ + blockIdx.x * 16;
  int col  = lane & 15;
  int radd = (lane < 16) ? 0 : 8;
  int arow = m0 + (lane & 15);        // A row this lane reads from LDS
  int ahi8 = ((lane >> 4) & 1) * 8;   // K sub-offset per ISA 16-bit A layout
  int crow = threadIdx.x >> 3;        // copy: 8 threads per 512-bf16 row
  int csub = threadIdx.x & 7;

  // Register-resident recurrent weights: 16 fragments = this wave's 16x512 strip
  v16bf w[16];
#pragma unroll
  for (int kk = 0; kk < 16; ++kk)
    w[kk] = load_frag(Wh, H_, n0, 1 << 28, kk * 32, lane);

  for (int t = 0; t < T_; ++t) {
    const bf16_t* hin = hbuf + ((size_t)(t & 1) * 2 + chain) * (B_ * H_);
    bf16_t* hout      = hbuf + ((size_t)((t + 1) & 1) * 2 + chain) * (B_ * H_);

    // Stage h_prev into LDS: all 8 loads live at once -> single batched wait
    {
      const uint4* src = (const uint4*)hin;  // 32 rows x 64 uint4
      uint4 tmp[8];
#pragma unroll
      for (int j = 0; j < 8; ++j)
        tmp[j] = src[crow * 64 + csub + j * 8];
#pragma unroll
      for (int j = 0; j < 8; ++j)
        *(uint4*)&sh_h[crow * LDH + (csub + j * 8) * 8] = tmp[j];
    }
    __syncthreads();

    v8f acc = {};
#pragma unroll
    for (int kk = 0; kk < 16; ++kk) {
      const bf16_t* p = &sh_h[arow * LDH + kk * 32 + ahi8];
      v8bf lo = *(const v8bf*)p;
      v8bf hi = *(const v8bf*)(p + 16);
      v16bf a = __builtin_shufflevector(lo, hi,
                  0,1,2,3,4,5,6,7,8,9,10,11,12,13,14,15);
      acc = wmma_bf16(a, w[kk], acc);
    }
    for (int r = 0; r < 8; ++r)
      lds[wave * 256 + (r + radd) * 16 + col] = acc[r];
    __syncthreads();

    for (int e = threadIdx.x; e < 512; e += 256) {
      int b = e >> 4, cc = e & 15;
      int mtt = b >> 4, ri = b & 15;
      int nh = blockIdx.x * 16 + cc;
      size_t gxr = ((size_t)b * T_ + t) * G4_;
      float gi = lds[(mtt * 4 + 0) * 256 + ri * 16 + cc] + Gx[gxr + 0 * H_ + nh];
      float gf = lds[(mtt * 4 + 1) * 256 + ri * 16 + cc] + Gx[gxr + 1 * H_ + nh];
      float gg = lds[(mtt * 4 + 2) * 256 + ri * 16 + cc] + Gx[gxr + 2 * H_ + nh];
      float go = lds[(mtt * 4 + 3) * 256 + ri * 16 + cc] + Gx[gxr + 3 * H_ + nh];
      float si = 1.f / (1.f + expf(-gi));
      float sf = 1.f / (1.f + expf(-gf));
      float so = 1.f / (1.f + expf(-go));
      float tg = tanhf(gg);
      int ci = b * H_ + nh;
      float cn = sf * cs[ci] + si * tg;
      float hn = so * tanhf(cn);
      cs[ci]   = cn;
      hout[ci] = (bf16_t)hn;
      outp[((size_t)b * T_ + t) * H_ + nh] = hn;
      if (chain == 0) out_e_bf[((size_t)b * T_ + t) * H_ + nh] = (bf16_t)hn;
    }

    // per-chain device-scope inter-step barrier (release writes, arrive, spin)
    __threadfence();
    __syncthreads();
    if (threadIdx.x == 0) {
      atomicAdd(&cbar[t], 1u);
      while (__hip_atomic_load(&cbar[t], __ATOMIC_ACQUIRE,
                               __HIP_MEMORY_SCOPE_AGENT) < NBLK_CHAIN)
        __builtin_amdgcn_s_sleep(1);
    }
    __syncthreads();
  }
}

// ---------------------------------------------------------------------------
// Small support kernels
// ---------------------------------------------------------------------------
__global__ void init_state_kernel(const float* __restrict__ h0,
                                  const float* __restrict__ c0,
                                  bf16_t* __restrict__ hbuf,
                                  float* __restrict__ cstate,
                                  unsigned* __restrict__ bar) {
  int i = blockIdx.x * blockDim.x + threadIdx.x;
  if (i < B_ * H_) {
    bf16_t h = (bf16_t)h0[i];
    hbuf[i] = h;                       // buf0 chain e
    hbuf[B_ * H_ + i] = h;             // buf0 chain w
    hbuf[2 * B_ * H_ + i] = (bf16_t)0.f;
    hbuf[3 * B_ * H_ + i] = (bf16_t)0.f;
    cstate[i] = c0[i];
    cstate[B_ * H_ + i] = c0[i];
  }
  if (i < 2 * T_) bar[i] = 0u;         // re-arm barrier counters every call
}

__global__ void embed_gather_kernel(const int* __restrict__ tok,
                                    const float* __restrict__ emb,
                                    bf16_t* __restrict__ out, int ntok) {
  int i = blockIdx.x * blockDim.x + threadIdx.x;
  if (i < ntok * E_) {
    int tkn = tok[i >> 8];
    out[i] = (bf16_t)emb[(size_t)tkn * E_ + (i & 255)];
  }
}

__global__ void conv_bf16_kernel(const float* __restrict__ src,
                                 bf16_t* __restrict__ dst, int n) {
  int i = blockIdx.x * blockDim.x + threadIdx.x;
  if (i < n) dst[i] = (bf16_t)src[i];
}

__global__ void conv_emb_pad_kernel(const float* __restrict__ emb,
                                    bf16_t* __restrict__ dst) {
  int i = blockIdx.x * blockDim.x + threadIdx.x;
  if (i < VP_ * E_) dst[i] = (i < V_ * E_) ? (bf16_t)emb[i] : (bf16_t)0.f;
}

__global__ void bias_sum_kernel(const float* bi_e, const float* bh_e,
                                const float* bi_w, const float* bh_w,
                                float* be, float* bw) {
  int i = blockIdx.x * blockDim.x + threadIdx.x;
  if (i < G4_) { be[i] = bi_e[i] + bh_e[i]; bw[i] = bi_w[i] + bh_w[i]; }
}

__global__ __launch_bounds__(256) void attention_kernel(
    const float* __restrict__ key, const float* __restrict__ enc,
    float* __restrict__ applied) {
  __shared__ float skey[H_];
  __shared__ float sc[S_];
  __shared__ float sred;
  int bt = blockIdx.x;
  int b  = bt / T_;
  const float* kp = key + (size_t)bt * H_;
  const float* ep = enc + (size_t)b * S_ * H_;
  for (int k = threadIdx.x; k < H_; k += 256) skey[k] = kp[k];
  __syncthreads();
  int wid = threadIdx.x >> 5, lane = threadIdx.x & 31;
  for (int s = wid; s < S_; s += 8) {
    float sum = 0.f;
    const float* er = ep + (size_t)s * H_;
    for (int k = lane; k < H_; k += 32) sum += skey[k] * er[k];
    for (int off = 16; off > 0; off >>= 1) sum += __shfl_xor(sum, off, 32);
    if (lane == 0) sc[s] = sum;
  }
  __syncthreads();
  if (threadIdx.x == 0) {
    float m = sc[0];
    for (int s = 1; s < S_; ++s) m = fmaxf(m, sc[s]);
    sred = m;
  }
  __syncthreads();
  if (threadIdx.x < S_) sc[threadIdx.x] = expf(sc[threadIdx.x] - sred);
  __syncthreads();
  if (threadIdx.x == 0) {
    float sum = 0.f;
    for (int s = 0; s < S_; ++s) sum += sc[s];
    sred = 1.f / sum;
  }
  __syncthreads();
  if (threadIdx.x < S_) sc[threadIdx.x] *= sred;
  __syncthreads();
  for (int k = threadIdx.x; k < H_; k += 256) {
    float a = 0.f;
    for (int s = 0; s < S_; ++s) a += sc[s] * ep[(size_t)s * H_ + k];
    applied[(size_t)bt * H_ + k] = a;
  }
}

__global__ void counters_kernel(const int* __restrict__ edits,
                                int* __restrict__ ctrd, int* __restrict__ ctri) {
  int b = threadIdx.x;
  if (b < B_) {
    int cd = 0, ci = 0;
    for (int j = 0; j < T_ - 1; ++j) {
      int a = edits[b * T_ + 1 + j];
      ctrd[b * (T_ - 1) + j] = (cd < S_ - 1) ? cd : (S_ - 1);
      ctri[b * (T_ - 1) + j] = (ci < S_ - 1) ? ci : (S_ - 1);
      cd += (a == 2 || a == 3) ? 1 : 0;                 // KEEP | DEL
      ci += (a != 3 && a != 5 && a != 0) ? 1 : 0;       // !DEL & !STOP & !PAD
    }
  }
}

__global__ void build_feat_kernel(const float* __restrict__ out_e,
                                  const float* __restrict__ applied,
                                  const float* __restrict__ enc,
                                  const float* __restrict__ out_w,
                                  const int* __restrict__ ctrd,
                                  const int* __restrict__ ctri,
                                  bf16_t* __restrict__ feat) {
  int i = blockIdx.x * 256 + threadIdx.x;
  const int total = B_ * (T_ - 1) * G4_;
  if (i < total) {
    int col = i & (G4_ - 1);
    int row = i >> 11;
    int b = row / (T_ - 1), tp = row % (T_ - 1);
    float v;
    if (col < H_)
      v = out_e[((size_t)b * T_ + tp) * H_ + col];
    else if (col < 2 * H_)
      v = applied[((size_t)b * T_ + tp) * H_ + (col - H_)];
    else if (col < 3 * H_)
      v = enc[((size_t)b * S_ + ctrd[b * (T_ - 1) + tp]) * H_ + (col - 2 * H_)];
    else
      v = out_w[((size_t)b * T_ + ctri[b * (T_ - 1) + tp]) * H_ + (col - 3 * H_)];
    feat[i] = (bf16_t)v;
  }
}

__global__ __launch_bounds__(256) void logsoftmax_kernel(float* __restrict__ x) {
  __shared__ float red[256];
  float* p = x + (size_t)blockIdx.x * V_;
  float m = -3.4e38f;
  for (int k = threadIdx.x; k < V_; k += 256) m = fmaxf(m, p[k]);
  red[threadIdx.x] = m; __syncthreads();
  for (int s = 128; s > 0; s >>= 1) {
    if (threadIdx.x < s) red[threadIdx.x] = fmaxf(red[threadIdx.x], red[threadIdx.x + s]);
    __syncthreads();
  }
  m = red[0]; __syncthreads();
  float sum = 0.f;
  for (int k = threadIdx.x; k < V_; k += 256) sum += expf(p[k] - m);
  red[threadIdx.x] = sum; __syncthreads();
  for (int s = 128; s > 0; s >>= 1) {
    if (threadIdx.x < s) red[threadIdx.x] += red[threadIdx.x + s];
    __syncthreads();
  }
  float l = m + logf(red[0]);
  for (int k = threadIdx.x; k < V_; k += 256) p[k] = p[k] - l;
}

__global__ void finalize_hc_kernel(const float* __restrict__ out_e,
                                   const float* __restrict__ cstate,
                                   float* __restrict__ dout) {
  int i = blockIdx.x * blockDim.x + threadIdx.x;
  if (i < B_ * H_) {
    int b = i >> 9, nh = i & (H_ - 1);
    size_t base = (size_t)B_ * (T_ - 1) * V_;  // end of logp
    dout[base + i]           = out_e[((size_t)b * T_ + (T_ - 1)) * H_ + nh];
    dout[base + B_ * H_ + i] = cstate[i];      // chain 0 = edit LSTM
  }
}

// ---------------------------------------------------------------------------
// Host launcher
// ---------------------------------------------------------------------------
extern "C" void kernel_launch(void* const* d_in, const int* in_sizes, int n_in,
                              void* d_out, int out_size, void* d_ws, size_t ws_size,
                              hipStream_t stream) {
  (void)in_sizes; (void)n_in; (void)out_size; (void)ws_size;
  const int*   input_edits = (const int*)d_in[0];
  const int*   simp_sent   = (const int*)d_in[1];
  const float* enc   = (const float*)d_in[3];
  const float* h0    = (const float*)d_in[4];
  const float* c0    = (const float*)d_in[5];
  const float* emb   = (const float*)d_in[6];
  const float* Wi_e  = (const float*)d_in[7];
  const float* Wh_e  = (const float*)d_in[8];
  const float* bi_e  = (const float*)d_in[9];
  const float* bh_e  = (const float*)d_in[10];
  const float* Wi_w  = (const float*)d_in[11];
  const float* Wh_w  = (const float*)d_in[12];
  const float* bi_w  = (const float*)d_in[13];
  const float* bh_w  = (const float*)d_in[14];
  const float* Wp    = (const float*)d_in[15];
  const float* Wm    = (const float*)d_in[16];
  const float* bm    = (const float*)d_in[17];
  const float* b_out = (const float*)d_in[18];
  float* out = (float*)d_out;

  char* base = (char*)d_ws; size_t off = 0;
  auto alloc = [&](size_t bytes) -> void* {
    off = (off + 255) & ~(size_t)255;
    void* p = base + off; off += bytes; return p;
  };

  bf16_t*   x_e_b  = (bf16_t*)alloc((size_t)B_ * T_ * E_ * 2);
  bf16_t*   x_w_b  = (bf16_t*)alloc((size_t)B_ * S_ * E_ * 2);
  bf16_t*   Wi_e_b = (bf16_t*)alloc((size_t)G4_ * E_ * 2);
  bf16_t*   Wh_e_b = (bf16_t*)alloc((size_t)G4_ * H_ * 2);
  bf16_t*   Wi_w_b = (bf16_t*)alloc((size_t)G4_ * E_ * 2);
  bf16_t*   Wh_w_b = (bf16_t*)alloc((size_t)G4_ * H_ * 2);
  bf16_t*   Wp_b   = (bf16_t*)alloc((size_t)H_ * H_ * 2);
  bf16_t*   Wm_b   = (bf16_t*)alloc((size_t)E_ * G4_ * 2);
  bf16_t*   emb_b  = (bf16_t*)alloc((size_t)VP_ * E_ * 2);
  float*    bse    = (float*)alloc((size_t)G4_ * 4);
  float*    bsw    = (float*)alloc((size_t)G4_ * 4);
  float*    Gx_e   = (float*)alloc((size_t)B_ * T_ * G4_ * 4);
  float*    Gx_w   = (float*)alloc((size_t)B_ * S_ * G4_ * 4);
  float*    out_e  = (float*)alloc((size_t)B_ * T_ * H_ * 4);
  float*    out_w  = (float*)alloc((size_t)B_ * S_ * H_ * 4);
  bf16_t*   out_e_b= (bf16_t*)alloc((size_t)B_ * T_ * H_ * 2);
  bf16_t*   hbuf   = (bf16_t*)alloc((size_t)4 * B_ * H_ * 2);
  float*    cstate = (float*)alloc((size_t)2 * B_ * H_ * 4);
  float*    keyb   = (float*)alloc((size_t)B_ * T_ * H_ * 4);
  float*    applied= (float*)alloc((size_t)B_ * T_ * H_ * 4);
  bf16_t*   feat   = (bf16_t*)alloc((size_t)B_ * (T_-1) * G4_ * 2);
  bf16_t*   hmlp   = (bf16_t*)alloc((size_t)B_ * (T_-1) * E_ * 2);
  int*      ctrd   = (int*)alloc((size_t)B_ * (T_-1) * 4);
  int*      ctri   = (int*)alloc((size_t)B_ * (T_-1) * 4);
  unsigned* bar    = (unsigned*)alloc((size_t)2 * T_ * 4);

  // 1) Embedding gathers (fp32 -> bf16)
  embed_gather_kernel<<<(B_*T_*E_+255)/256, 256, 0, stream>>>(input_edits, emb, x_e_b, B_*T_);
  embed_gather_kernel<<<(B_*S_*E_+255)/256, 256, 0, stream>>>(simp_sent,   emb, x_w_b, B_*S_);

  // 2) Weight conversions + state/barrier init
  conv_bf16_kernel<<<(G4_*E_+255)/256, 256, 0, stream>>>(Wi_e, Wi_e_b, G4_*E_);
  conv_bf16_kernel<<<(G4_*H_+255)/256, 256, 0, stream>>>(Wh_e, Wh_e_b, G4_*H_);
  conv_bf16_kernel<<<(G4_*E_+255)/256, 256, 0, stream>>>(Wi_w, Wi_w_b, G4_*E_);
  conv_bf16_kernel<<<(G4_*H_+255)/256, 256, 0, stream>>>(Wh_w, Wh_w_b, G4_*H_);
  conv_bf16_kernel<<<(H_*H_+255)/256, 256, 0, stream>>>(Wp, Wp_b, H_*H_);
  conv_bf16_kernel<<<(E_*G4_+255)/256, 256, 0, stream>>>(Wm, Wm_b, E_*G4_);
  conv_emb_pad_kernel<<<(VP_*E_+255)/256, 256, 0, stream>>>(emb, emb_b);
  bias_sum_kernel<<<(G4_+255)/256, 256, 0, stream>>>(bi_e, bh_e, bi_w, bh_w, bse, bsw);
  init_state_kernel<<<(B_*H_+255)/256, 256, 0, stream>>>(h0, c0, hbuf, cstate, bar);

  // 3) Hoisted input-side gate preactivations: Gx = x*Wi^T + (bi+bh)  [WMMA]
  gemm_bf16_kernel<true,false,true,false><<<dim3(G4_/128, (B_*T_)/128), 256, 0, stream>>>(
      x_e_b, Wi_e_b, Gx_e, nullptr, bse, B_*T_, E_, G4_, G4_);
  gemm_bf16_kernel<true,false,true,false><<<dim3(G4_/128, (B_*S_)/128), 256, 0, stream>>>(
      x_w_b, Wi_w_b, Gx_w, nullptr, bsw, B_*S_, E_, G4_, G4_);

  // 4) Full recurrence, both chains, ONE persistent launch  [WMMA]
  lstm_persistent_kernel<<<dim3(H_/16, 2), 256, 0, stream>>>(
      Wh_e_b, Wh_w_b, Gx_e, Gx_w, hbuf, cstate, out_e, out_w, out_e_b, bar);

  // 5) key = out_e * Wp^T  [WMMA]
  gemm_bf16_kernel<false,false,true,false><<<dim3(H_/128, (B_*T_)/128), 256, 0, stream>>>(
      out_e_b, Wp_b, keyb, nullptr, nullptr, B_*T_, H_, H_, H_);

  // 6) attention softmax + apply
  attention_kernel<<<B_*T_, 256, 0, stream>>>(keyb, enc, applied);

  // 7) edit counters + feature concat (bf16)
  counters_kernel<<<1, 32, 0, stream>>>(input_edits, ctrd, ctri);
  build_feat_kernel<<<(B_*(T_-1)*G4_+255)/256, 256, 0, stream>>>(
      out_e, applied, enc, out_w, ctrd, ctri, feat);

  // 8) hmlp = tanh(feat * Wm^T + bm)  [WMMA, bf16 out]
  gemm_bf16_kernel<true,true,false,true><<<dim3(E_/128, (B_*(T_-1)+127)/128), 256, 0, stream>>>(
      feat, Wm_b, nullptr, hmlp, bm, B_*(T_-1), G4_, E_, E_);

  // 9) logits = hmlp * emb^T + b_out, straight into d_out  [WMMA]
  gemm_bf16_kernel<true,false,true,false><<<dim3(VP_/128, (B_*(T_-1)+127)/128), 256, 0, stream>>>(
      hmlp, emb_b, out, nullptr, b_out, B_*(T_-1), E_, V_, V_);

  // 10) in-place log-softmax over V, then he/ce tail
  logsoftmax_kernel<<<B_*(T_-1), 256, 0, stream>>>(out);
  finalize_hc_kernel<<<(B_*H_+255)/256, 256, 0, stream>>>(out_e, cstate, out);
}
